// my_random_walk_75900662055249
// MI455X (gfx1250) — matrix operand order repaired and checked by reading, not compiled, not run on hardware
//
#include <hip/hip_runtime.h>

// ---------------------------------------------------------------------------
// Problem constants (match reference)
// ---------------------------------------------------------------------------
#define D_IN    100
#define K1P     128   // D_IN padded to WMMA K granularity
#define D_HID   512
#define D_OUT   200
#define D_OUTP  208   // D_OUT padded to 16
#define WALKS   5
#define MAXLEN  4
#define ENTRIES (WALKS * MAXLEN)   // 20
#define NPATHF  10
#define M_BLK   32    // rows per GEMM block (2 row-tiles)

typedef __attribute__((ext_vector_type(16))) __bf16 v16bf;
typedef __attribute__((ext_vector_type(8)))  float  v8f;
typedef __attribute__((ext_vector_type(4)))  int    v4ia;   // async-copy unit

#define AS1 __attribute__((address_space(1)))
#define AS3 __attribute__((address_space(3)))

union Frag16 {
  v16bf v;
  uint4 q[2];
};

// ---------------------------------------------------------------------------
// Async global->LDS staging (gfx1250 GLOBAL_LOAD_ASYNC_TO_LDS_B128, ASYNCcnt)
// Builtin signature (from hipcc diagnostic): param0 = int4-vector pointer in
// addrspace(1) (printed '__device__'), so cast via AS1 void* -> AS1 v4i*.
// Guarded: falls back to register-staged uint4 copies if builtin missing.
// ---------------------------------------------------------------------------
#if defined(__has_builtin)
#if __has_builtin(__builtin_amdgcn_global_load_async_to_lds_b128) && \
    __has_builtin(__builtin_amdgcn_s_wait_asynccnt)
#define USE_ASYNC_LDS 1
#endif
#endif

__device__ __forceinline__ void copy16_to_lds(const unsigned short* g,
                                              unsigned short* l) {
#ifdef USE_ASYNC_LDS
  __builtin_amdgcn_global_load_async_to_lds_b128(
      (AS1 v4ia*)(const AS1 void*)g,
      (AS3 v4ia*)(AS3 void*)l, 0, 0);
#else
  *(uint4*)l = *(const uint4*)g;
#endif
}
// Stage one 64-byte row (4 x b128). Fallback loads all 4 before storing so the
// compiler can overlap the global loads instead of ping-ponging one register.
__device__ __forceinline__ void stage_row64(const unsigned short* g,
                                            unsigned short* l) {
#ifdef USE_ASYNC_LDS
  copy16_to_lds(g, l);
  copy16_to_lds(g + 8, l + 8);
  copy16_to_lds(g + 16, l + 16);
  copy16_to_lds(g + 24, l + 24);
#else
  const uint4* s = (const uint4*)g;
  uint4 a0 = s[0], a1 = s[1], a2 = s[2], a3 = s[3];
  uint4* d = (uint4*)l;
  d[0] = a0; d[1] = a1; d[2] = a2; d[3] = a3;
#endif
}
__device__ __forceinline__ void async_join() {
#ifdef USE_ASYNC_LDS
  __builtin_amdgcn_s_wait_asynccnt(0);
#endif
}

// ---------------------------------------------------------------------------
// bf16 helpers (RNE convert, bit-level)
// ---------------------------------------------------------------------------
__device__ __forceinline__ unsigned short f2bf(float f) {
  unsigned u = __float_as_uint(f);
  unsigned r = (u + 0x7FFFu + ((u >> 16) & 1u)) >> 16;
  return (unsigned short)r;
}
__device__ __forceinline__ float bf2f(unsigned us) {
  return __uint_as_float(us << 16);
}
__device__ __forceinline__ unsigned pack2(float a, float b) {
  return (unsigned)f2bf(a) | ((unsigned)f2bf(b) << 16);
}
__device__ __forceinline__ void add_bf8(float* acc, uint4 v) {
  acc[0] += bf2f(v.x & 0xffffu); acc[1] += bf2f(v.x >> 16);
  acc[2] += bf2f(v.y & 0xffffu); acc[3] += bf2f(v.y >> 16);
  acc[4] += bf2f(v.z & 0xffffu); acc[5] += bf2f(v.z >> 16);
  acc[6] += bf2f(v.w & 0xffffu); acc[7] += bf2f(v.w >> 16);
}

// ---------------------------------------------------------------------------
// Threefry-2x32 (JAX-compatible structure: 20 rounds, key inject every 4)
// ---------------------------------------------------------------------------
__device__ __forceinline__ void tf2x32(unsigned k0, unsigned k1,
                                       unsigned c0, unsigned c1,
                                       unsigned& o0, unsigned& o1) {
  unsigned ks2 = k0 ^ k1 ^ 0x1BD11BDAu;
  unsigned x0 = c0 + k0, x1 = c1 + k1;
#define TFR(r) { x0 += x1; x1 = (x1 << (r)) | (x1 >> (32 - (r))); x1 ^= x0; }
  TFR(13) TFR(15) TFR(26) TFR(6)   x0 += k1;  x1 += ks2 + 1u;
  TFR(17) TFR(29) TFR(16) TFR(24)  x0 += ks2; x1 += k0 + 2u;
  TFR(13) TFR(15) TFR(26) TFR(6)   x0 += k0;  x1 += k1 + 3u;
  TFR(17) TFR(29) TFR(16) TFR(24)  x0 += k1;  x1 += ks2 + 4u;
  TFR(13) TFR(15) TFR(26) TFR(6)   x0 += ks2; x1 += k0 + 5u;
#undef TFR
  o0 = x0; o1 = x1;
}

__device__ __forceinline__ unsigned randbits_even(unsigned k0, unsigned k1,
                                                  unsigned idx, unsigned M) {
  unsigned half = M >> 1;
  unsigned o0, o1;
  if (idx < half) { tf2x32(k0, k1, idx, idx + half, o0, o1); return o0; }
  tf2x32(k0, k1, idx - half, idx, o0, o1);
  return o1;
}
__device__ __forceinline__ float bits_to_unit(unsigned b) {
  return __uint_as_float((b >> 9) | 0x3F800000u) - 1.0f;
}

// ---------------------------------------------------------------------------
// Conversion kernels: f32 -> K-padded bf16 tiles
// ---------------------------------------------------------------------------
__global__ __launch_bounds__(256) void convert_x_kernel(
    const float* __restrict__ x, unsigned short* __restrict__ xb, int n) {
  int idx = blockIdx.x * 256 + threadIdx.x;
  if (idx >= n * K1P) return;
  int m = idx / K1P, k = idx - m * K1P;
  float v = (k < D_IN) ? x[(size_t)m * D_IN + k] : 0.0f;
  xb[idx] = f2bf(v);
}

__global__ __launch_bounds__(256) void convert_w1_kernel(
    const float* __restrict__ W1, unsigned short* __restrict__ w1b) {
  int idx = blockIdx.x * 256 + threadIdx.x;   // D_HID * K1P threads
  int o = idx / K1P, k = idx - o * K1P;
  float v = (k < D_IN) ? W1[(size_t)o * D_IN + k] : 0.0f;
  w1b[idx] = f2bf(v);
}

__global__ __launch_bounds__(256) void convert_w3_kernel(
    const float* __restrict__ W3, unsigned short* __restrict__ w3b) {
  int idx = blockIdx.x * 256 + threadIdx.x;   // D_OUTP * D_HID threads
  int o = idx / D_HID, k = idx - o * D_HID;
  float v = (o < D_OUT) ? W3[(size_t)o * D_HID + k] : 0.0f;
  w3b[idx] = f2bf(v);
}

// ---------------------------------------------------------------------------
// GEMM1: h[n,512] = relu(xb[n,128] @ w1b[512,128]^T + b1), bf16 out, f32 acc
// Block: 32 rows x 512 cols, 8 waves. wave -> row-tile (wave>>2), col group
// (wave&3): 8 col-tiles per wave, acc[8]. Async-staged LDS tiles.
// ---------------------------------------------------------------------------
__global__ __launch_bounds__(256) void gemm1_kernel(
    const unsigned short* __restrict__ xb, const unsigned short* __restrict__ w1b,
    const float* __restrict__ b1, unsigned short* __restrict__ h, int n) {
  __shared__ alignas(16) unsigned short Ash[M_BLK * 32];   // 2 KB
  __shared__ alignas(16) unsigned short Bsh[D_HID * 32];   // 32 KB
  const int tid = threadIdx.x, wave = tid >> 5, lane = tid & 31;
  const int m0 = blockIdx.x * M_BLK;
  const int rt = wave >> 2;        // row tile 0/1
  const int cw = wave & 3;         // col group

  v8f acc[8];
  for (int t = 0; t < 8; ++t)
    for (int i = 0; i < 8; ++i) acc[t][i] = 0.0f;

  for (int kc = 0; kc < K1P; kc += 32) {
    // Stage A tile: 32 rows x 32 K (2 KB); 128 threads x 16B
    if (tid < 128) {
      int r = tid >> 2, c = (tid & 3) * 8;
      int mr = m0 + r; if (mr >= n) mr = n - 1;
      copy16_to_lds(&xb[(size_t)mr * K1P + kc + c], &Ash[r * 32 + c]);
    }
    // Stage B tile: 512 cols x 32 K (32 KB), o-major (contiguous K per col)
    for (int r = tid; r < D_HID; r += 256)
      stage_row64(&w1b[(size_t)r * K1P + kc], &Bsh[r * 32]);
    async_join();
    __syncthreads();

    // A fragment (ISA 16-bit A 16x32 layout)
    Frag16 A;
    {
      int mrow = rt * 16 + (lane & 15), ko = (lane >= 16) ? 8 : 0;
      A.q[0] = *(const uint4*)&Ash[mrow * 32 + ko];
      A.q[1] = *(const uint4*)&Ash[mrow * 32 + ko + 16];
    }
    for (int t = 0; t < 8; ++t) {
      int o = (t * 4 + cw) * 16 + (lane & 15);
      int kb = (lane >= 16) ? 8 : 0;   // ISA 16-bit B 32x16 layout
      Frag16 B;
      B.q[0] = *(const uint4*)&Bsh[o * 32 + kb];
      B.q[1] = *(const uint4*)&Bsh[o * 32 + kb + 16];
      acc[t] = __builtin_amdgcn_wmma_f32_16x16x32_bf16(
          false, A.v, false, B.v, (short)0, acc[t], false, false);
    }
    __syncthreads();
  }

  // Epilogue: bias + relu, store bf16 h (halves gather traffic later)
  for (int t = 0; t < 8; ++t) {
    int o = (t * 4 + cw) * 16 + (lane & 15);
    float bias = b1[o];
    int mb = m0 + rt * 16 + ((lane >= 16) ? 8 : 0);
    for (int r = 0; r < 8; ++r) {
      int m = mb + r;
      float v = acc[t][r] + bias;
      v = v > 0.0f ? v : 0.0f;
      if (m < n) h[(size_t)m * D_HID + o] = f2bf(v);
    }
  }
}

// ---------------------------------------------------------------------------
// Fused random walks + neighborhood aggregation (one wave per node).
// ---------------------------------------------------------------------------
__global__ __launch_bounds__(256) void walk_agg_kernel(
    const unsigned short* __restrict__ h, const int* __restrict__ nbr,
    const int* __restrict__ deg, int maxd,
    unsigned short* __restrict__ aggb, int n) {
  __shared__ int ent[8][ENTRIES];
  __shared__ unsigned msk[8];
  const int tid = threadIdx.x, wave = tid >> 5, lane = tid & 31;
  const int node = blockIdx.x * 8 + wave;
  const bool active = node < n;

  if (lane == 0) msk[wave] = 0u;
  __syncthreads();

  if (active && lane < WALKS) {
    const int w = lane;
    int cur = node;
    bool valid = true;
    unsigned bits = 1u << w;            // entry e = step*5 + w; step 0 valid
    ent[wave][w] = cur;
    unsigned kA0, kA1;
    tf2x32(0u, 42u, 0u, 0u, kA0, kA1);  // fold_in(key(42), 0)
    for (int j = 0; j < MAXLEN - 1; ++j) {
      unsigned kj0, kj1;
      tf2x32(kA0, kA1, 0u, (unsigned)j, kj0, kj1);   // fold_in per step
      unsigned idx = (unsigned)w * (unsigned)n + (unsigned)node;
      float u = bits_to_unit(randbits_even(kj0, kj1, idx,
                                           (unsigned)(WALKS) * (unsigned)n));
      int d = deg[cur];
      int choice = (int)(u * (float)d);
      int dm1 = d - 1; if (dm1 < 0) dm1 = 0;
      if (choice > dm1) choice = dm1;
      valid = valid && (d > 0);
      if (valid) cur = nbr[(size_t)cur * maxd + choice];
      ent[wave][(j + 1) * WALKS + w] = cur;
      if (valid) bits |= 1u << ((j + 1) * WALKS + w);
    }
    atomicOr(&msk[wave], bits);
  }
  __syncthreads();

  float acc[16];
  for (int i = 0; i < 16; ++i) acc[i] = 0.0f;
  unsigned m = msk[wave];
  int cnt = __popc(m);

  if (active) {
    for (int e = 0; e < ENTRIES; ++e) {
      if (e + 1 < ENTRIES && ((m >> (e + 1)) & 1u))
        __builtin_prefetch(
            &h[(size_t)ent[wave][e + 1] * D_HID + lane * 16], 0, 0);
      if ((m >> e) & 1u) {
        const uint4* p =
            (const uint4*)&h[(size_t)ent[wave][e] * D_HID + lane * 16];
        uint4 a = p[0], b = p[1];
        add_bf8(acc, a);
        add_bf8(acc + 8, b);
      }
    }
    float inv = 1.0f / (float)cnt;
    uint4 q0, q1;
    q0.x = pack2(acc[0] * inv, acc[1] * inv);
    q0.y = pack2(acc[2] * inv, acc[3] * inv);
    q0.z = pack2(acc[4] * inv, acc[5] * inv);
    q0.w = pack2(acc[6] * inv, acc[7] * inv);
    q1.x = pack2(acc[8] * inv, acc[9] * inv);
    q1.y = pack2(acc[10] * inv, acc[11] * inv);
    q1.z = pack2(acc[12] * inv, acc[13] * inv);
    q1.w = pack2(acc[14] * inv, acc[15] * inv);
    uint4* dst = (uint4*)&aggb[(size_t)node * D_HID + lane * 16];
    dst[0] = q0; dst[1] = q1;
  }
}

// ---------------------------------------------------------------------------
// GEMM2: out2[n,200] = relu(aggb[n,512] @ w3b[208,512]^T + b3), f32 out
// Block: 32 rows x 208 cols; 26 flat tiles striped across 8 waves, acc[4].
// ---------------------------------------------------------------------------
__global__ __launch_bounds__(256) void gemm2_kernel(
    const unsigned short* __restrict__ aggb, const unsigned short* __restrict__ w3b,
    const float* __restrict__ b3, float* __restrict__ out2, int n) {
  __shared__ alignas(16) unsigned short Ash[M_BLK * 32];    // 2 KB
  __shared__ alignas(16) unsigned short Bsh[D_OUTP * 32];   // 13 KB
  const int tid = threadIdx.x, wave = tid >> 5, lane = tid & 31;
  const int m0 = blockIdx.x * M_BLK;
  const int ntiles = D_OUTP / 16;          // 13
  const int nflat = 2 * ntiles;            // 26

  v8f acc[4];
  for (int t = 0; t < 4; ++t)
    for (int i = 0; i < 8; ++i) acc[t][i] = 0.0f;

  for (int kc = 0; kc < D_HID; kc += 32) {
    if (tid < 128) {
      int r = tid >> 2, c = (tid & 3) * 8;
      int mr = m0 + r; if (mr >= n) mr = n - 1;
      copy16_to_lds(&aggb[(size_t)mr * D_HID + kc + c], &Ash[r * 32 + c]);
    }
    for (int r = tid; r < D_OUTP; r += 256)
      stage_row64(&w3b[(size_t)r * D_HID + kc], &Bsh[r * 32]);
    async_join();
    __syncthreads();

    // Both row-tile A fragments
    Frag16 A0, A1;
    {
      int mrow = lane & 15, ko = (lane >= 16) ? 8 : 0;
      A0.q[0] = *(const uint4*)&Ash[mrow * 32 + ko];
      A0.q[1] = *(const uint4*)&Ash[mrow * 32 + ko + 16];
      A1.q[0] = *(const uint4*)&Ash[(16 + mrow) * 32 + ko];
      A1.q[1] = *(const uint4*)&Ash[(16 + mrow) * 32 + ko + 16];
    }
    int t = 0;
    for (int ft = wave; ft < nflat; ft += 8, ++t) {
      int rtile = (ft >= ntiles) ? 1 : 0;
      int ot = ft - rtile * ntiles;
      int o = ot * 16 + (lane & 15);
      int kb = (lane >= 16) ? 8 : 0;
      Frag16 B;
      B.q[0] = *(const uint4*)&Bsh[o * 32 + kb];
      B.q[1] = *(const uint4*)&Bsh[o * 32 + kb + 16];
      acc[t] = __builtin_amdgcn_wmma_f32_16x16x32_bf16(
          false, (rtile ? A1.v : A0.v), false, B.v, (short)0, acc[t],
          false, false);
    }
    __syncthreads();
  }

  int t = 0;
  for (int ft = wave; ft < nflat; ft += 8, ++t) {
    int rtile = (ft >= ntiles) ? 1 : 0;
    int ot = ft - rtile * ntiles;
    int o = ot * 16 + (lane & 15);
    if (o < D_OUT) {
      float bias = b3[o];
      int mb = m0 + rtile * 16 + ((lane >= 16) ? 8 : 0);
      for (int r = 0; r < 8; ++r) {
        int m = mb + r;
        float v = acc[t][r] + bias;
        v = v > 0.0f ? v : 0.0f;
        if (m < n) out2[(size_t)m * D_OUT + o] = v;
      }
    }
  }
}

// ---------------------------------------------------------------------------
// Finalize: 10 directed walks, weighted mean of gathered out2 rows -> 200 f32
// ---------------------------------------------------------------------------
__global__ __launch_bounds__(256) void finalize_kernel(
    const float* __restrict__ out2, const int* __restrict__ nbr,
    const int* __restrict__ deg, int maxd, float* __restrict__ out, int n) {
  __shared__ int fn[MAXLEN * NPATHF];
  __shared__ float fw[MAXLEN * NPATHF];
  __shared__ float wsum;
  const int tid = threadIdx.x;

  if (tid < NPATHF) {
    unsigned kS0, kS1;
    tf2x32(0u, 42u, 0u, 1u, kS0, kS1);            // fold_in(key(42), 1)
    unsigned b = randbits_even(kS0, kS1, (unsigned)tid, (unsigned)NPATHF);
    int cur = (int)(b % (unsigned)n);
    fn[tid] = cur;
    fw[tid] = 1.0f;
    unsigned kW0, kW1;
    tf2x32(0u, 42u, 0u, 2u, kW0, kW1);            // fold_in(key(42), 2)
    bool valid = true;
    for (int j = 0; j < MAXLEN - 1; ++j) {
      unsigned kj0, kj1;
      tf2x32(kW0, kW1, 0u, (unsigned)j, kj0, kj1);
      float u = bits_to_unit(
          randbits_even(kj0, kj1, (unsigned)tid, (unsigned)NPATHF));
      int d = deg[cur];
      int choice = (int)(u * (float)d);
      int dm1 = d - 1; if (dm1 < 0) dm1 = 0;
      if (choice > dm1) choice = dm1;
      valid = valid && (d > 0);
      if (valid) cur = nbr[(size_t)cur * maxd + choice];
      fn[(j + 1) * NPATHF + tid] = cur;
      fw[(j + 1) * NPATHF + tid] = valid ? 1.0f : 0.0f;
    }
  }
  __syncthreads();
  if (tid == 0) {
    float s = 0.0f;
    for (int i = 0; i < MAXLEN * NPATHF; ++i) s += fw[i];
    wsum = s;
  }
  __syncthreads();
  if (tid < D_OUT) {
    float s = 0.0f;
    for (int e = 0; e < MAXLEN * NPATHF; ++e) {
      float w = fw[e];
      if (w != 0.0f) s += w * out2[(size_t)fn[e] * D_OUT + tid];
    }
    out[tid] = s / wsum;
  }
}

// ---------------------------------------------------------------------------
// Host entry
// ---------------------------------------------------------------------------
extern "C" void kernel_launch(void* const* d_in, const int* in_sizes, int n_in,
                              void* d_out, int out_size, void* d_ws, size_t ws_size,
                              hipStream_t stream) {
  (void)n_in; (void)out_size; (void)ws_size;
  const float* x      = (const float*)d_in[0];
  // d_in[1] = edge_index (unused; neighbor tables are precomputed)
  const int* nbr_bi   = (const int*)d_in[2];
  const int* deg_bi   = (const int*)d_in[3];
  const int* nbr_dir  = (const int*)d_in[4];
  const int* deg_dir  = (const int*)d_in[5];
  const float* W1     = (const float*)d_in[6];
  const float* b1     = (const float*)d_in[7];
  const float* W3     = (const float*)d_in[8];
  const float* b3     = (const float*)d_in[9];

  const int n = in_sizes[0] / D_IN;
  const int maxd_bi  = in_sizes[2] / n;
  const int maxd_dir = in_sizes[4] / n;

  char* ws = (char*)d_ws;
  auto carve = [&](size_t bytes) -> char* {
    char* p = ws;
    ws += (bytes + 255) & ~(size_t)255;
    return p;
  };
  unsigned short* xb   = (unsigned short*)carve((size_t)n * K1P * 2);
  unsigned short* h    = (unsigned short*)carve((size_t)n * D_HID * 2);
  unsigned short* aggb = (unsigned short*)carve((size_t)n * D_HID * 2);
  float*          out2 = (float*)carve((size_t)n * D_OUT * 4);
  unsigned short* w1b  = (unsigned short*)carve((size_t)D_HID * K1P * 2);
  unsigned short* w3b  = (unsigned short*)carve((size_t)D_OUTP * D_HID * 2);

  convert_x_kernel<<<((size_t)n * K1P + 255) / 256, 256, 0, stream>>>(x, xb, n);
  convert_w1_kernel<<<(D_HID * K1P) / 256, 256, 0, stream>>>(W1, w1b);
  convert_w3_kernel<<<(D_OUTP * D_HID) / 256, 256, 0, stream>>>(W3, w3b);

  gemm1_kernel<<<(n + M_BLK - 1) / M_BLK, 256, 0, stream>>>(xb, w1b, b1, h, n);
  walk_agg_kernel<<<(n + 7) / 8, 256, 0, stream>>>(h, nbr_bi, deg_bi, maxd_bi,
                                                   aggb, n);
  gemm2_kernel<<<(n + M_BLK - 1) / M_BLK, 256, 0, stream>>>(aggb, w3b, b3,
                                                            out2, n);
  finalize_kernel<<<1, 256, 0, stream>>>(out2, nbr_dir, deg_dir, maxd_dir,
                                         (float*)d_out, n);
}
